// BitNetTernaryLayer_47072841564532
// MI455X (gfx1250) — matrix operand order repaired and checked by reading, not compiled, not run on hardware
//
#include <hip/hip_runtime.h>
#include <hip/hip_bf16.h>

typedef __attribute__((ext_vector_type(16))) _Float16 v16h;
typedef __attribute__((ext_vector_type(4)))  _Float16 v4h;
typedef __attribute__((ext_vector_type(8)))  float    v8f;
typedef __attribute__((ext_vector_type(4)))  float    v4f;

#define M_DIM 16384
#define N_DIM 4096
#define K_DIM 4096
#define BM 128
#define BN 128
#define BK 32

// ---------------- Kernel 1: sum(|W|) ----------------
__global__ void bitnet_absum_kernel(const float* __restrict__ w,
                                    float* __restrict__ out, int n) {
    __shared__ float red[256];
    float s = 0.f;
    for (int i = blockIdx.x * blockDim.x + threadIdx.x; i < n;
         i += gridDim.x * blockDim.x)
        s += fabsf(w[i]);
    red[threadIdx.x] = s;
    __syncthreads();
    for (int off = 128; off > 0; off >>= 1) {
        if ((int)threadIdx.x < off) red[threadIdx.x] += red[threadIdx.x + off];
        __syncthreads();
    }
    if (threadIdx.x == 0) atomicAdd(out, red[0]);
}

// ---------------- Kernel 2: ternary-quantize W -> fp16, layout [N][K] ----------------
__global__ void bitnet_quantw_kernel(const float* __restrict__ w,
                                     const float* __restrict__ sum,
                                     _Float16* __restrict__ wq, int n4) {
    int i = blockIdx.x * blockDim.x + threadIdx.x;
    if (i >= n4) return;
    const float thr = 0.7f * (sum[0] * (1.0f / ((float)N_DIM * (float)K_DIM)));
    v4f v = ((const v4f*)w)[i];
    v4h q;
#pragma unroll
    for (int j = 0; j < 4; ++j) {
        float a = v[j];
        float t = (fabsf(a) > thr) ? ((a > 0.f) ? 1.f : -1.f) : 0.f;
        q[j] = (_Float16)t;
    }
    ((v4h*)wq)[i] = q;
}

// ---------------- Kernel 3: x fp32 -> fp16 ----------------
__global__ void bitnet_cvtx_kernel(const float* __restrict__ x,
                                   _Float16* __restrict__ xh, int n4) {
    int i = blockIdx.x * blockDim.x + threadIdx.x;
    if (i >= n4) return;
    v4f v = ((const v4f*)x)[i];
    v4h h;
#pragma unroll
    for (int j = 0; j < 4; ++j) h[j] = (_Float16)v[j];
    ((v4h*)xh)[i] = h;
}

// Async-DMA one 64-byte row (4 x b128) from global directly into LDS.
// IOFFSET is applied to BOTH the LDS destination and the global source
// (ISA 10.2 / 15.18.3), so one LDS base + one 64-bit address covers the row.
__device__ __forceinline__ void async_row64(unsigned lds, const _Float16* g) {
    asm volatile(
        "global_load_async_to_lds_b128 %0, %1, off\n\t"
        "global_load_async_to_lds_b128 %0, %1, off offset:16\n\t"
        "global_load_async_to_lds_b128 %0, %1, off offset:32\n\t"
        "global_load_async_to_lds_b128 %0, %1, off offset:48"
        :: "v"(lds), "v"(g) : "memory");
}

__device__ __forceinline__ void wait_async0() {
    asm volatile("s_wait_asynccnt 0x0" ::: "memory");
}

// ---------------- Kernel 4: WMMA GEMM C[M][N] = A[M][K] * B[N][K]^T ----------------
__global__ __launch_bounds__(128)
void bitnet_wmma_gemm_kernel(const _Float16* __restrict__ A,   // xh  [M][K]
                             const _Float16* __restrict__ B,   // wq  [N][K]
                             float* __restrict__ C) {          // out [M][N]
    __shared__ _Float16 sA[2][BM][BK];   // 16 KB (2 x 8 KB buffers)
    __shared__ _Float16 sB[2][BN][BK];   // 16 KB

    const int t    = threadIdx.x;
    const int lane = t & 31;             // wave32
    const int wave = t >> 5;             // 4 waves
    const int wr   = wave >> 1;          // wave row (0..1) -> 64 rows
    const int wc   = wave & 1;           // wave col (0..1) -> 64 cols
    const int m0   = blockIdx.y * BM;
    const int n0   = blockIdx.x * BN;

    // A fragment lane addressing (ISA 7.12.2, 16-bit A 16x32):
    //   lanes 0-15:  M=lane,    K chunks {0..7, 16..23}
    //   lanes 16-31: M=lane-16, K chunks {8..15, 24..31}
    const int am = lane & 15;
    const int ak = (lane >> 4) ? 8 : 0;
    // B fragment (32x16): lane holds column n=lane&15;
    //   lanes 0-15: K chunks {0..7, 16..23}; lanes 16-31: K chunks {8..15, 24..31}
    const int bn = lane & 15;
    const int bk = (lane >> 4) ? 8 : 0;

    v8f acc[4][4] = {};

    // Global staging: thread t owns row t of both 128x32 tiles (64 B each)
    const _Float16* gA = A + (size_t)(m0 + t) * K_DIM;
    const _Float16* gB = B + (size_t)(n0 + t) * K_DIM;

    // LDS byte addresses of this thread's destination rows (generic ptr
    // low 32 bits == LDS offset per ISA 10.2 aperture mapping)
    const unsigned ldsA0 = (unsigned)(unsigned long long)(const void*)&sA[0][t][0];
    const unsigned ldsB0 = (unsigned)(unsigned long long)(const void*)&sB[0][t][0];
    const unsigned bufA  = (unsigned)(BM * BK * sizeof(_Float16));   // 8192
    const unsigned bufB  = (unsigned)(BN * BK * sizeof(_Float16));   // 8192

    // Prologue: async-DMA tile 0 into buffer 0
    async_row64(ldsA0, gA);
    async_row64(ldsB0, gB);
    wait_async0();
    __syncthreads();

    const int KT = K_DIM / BK;           // 128 K-steps
    for (int kt = 0; kt < KT; ++kt) {
        const int cur = kt & 1;
        const int k1  = (kt + 1) * BK;

        if (kt + 1 < KT) {
            // async-DMA next K-tile into the other buffer (no VGPR staging)
            const unsigned nb = (unsigned)(cur ^ 1);
            async_row64(ldsA0 + nb * bufA, gA + k1);
            async_row64(ldsB0 + nb * bufB, gB + k1);
            if (kt + 2 < KT) {           // L2 prefetch two tiles ahead
                __builtin_prefetch(gA + k1 + BK, 0, 3);
                __builtin_prefetch(gB + k1 + BK, 0, 3);
            }
        }

        // Load 4 A + 4 B fragments from LDS (ds_load_b128 pairs)
        v16h af[4], bf[4];
#pragma unroll
        for (int i = 0; i < 4; ++i) {
            union { v16h v; v4f f[2]; } u;
            const _Float16* base = &sA[cur][wr * 64 + i * 16 + am][0];
            u.f[0] = *(const v4f*)(base + ak);
            u.f[1] = *(const v4f*)(base + ak + 16);
            af[i] = u.v;
        }
#pragma unroll
        for (int j = 0; j < 4; ++j) {
            union { v16h v; v4f f[2]; } u;
            const _Float16* base = &sB[cur][wc * 64 + j * 16 + bn][0];
            u.f[0] = *(const v4f*)(base + bk);
            u.f[1] = *(const v4f*)(base + bk + 16);
            bf[j] = u.v;
        }

        // 16 WMMAs per wave per K-step
#pragma unroll
        for (int i = 0; i < 4; ++i)
#pragma unroll
            for (int j = 0; j < 4; ++j)
                acc[i][j] = __builtin_amdgcn_wmma_f32_16x16x32_f16(
                    false, af[i], false, bf[j],
                    (short)0, acc[i][j], false, false);

        // Publish the async-filled buffer: drain ASYNCcnt, then barrier
        if (kt + 1 < KT) wait_async0();
        __syncthreads();
    }

    // C/D layout (ISA 7.12.2): VGPR r -> M = r + 8*(lane>=16), N = lane&15
    const int crow = (lane >> 4) * 8;
    const int ccol = lane & 15;
#pragma unroll
    for (int i = 0; i < 4; ++i)
#pragma unroll
        for (int j = 0; j < 4; ++j) {
            const size_t rbase = (size_t)(m0 + wr * 64 + i * 16 + crow) * N_DIM;
            const int    col   = n0 + wc * 64 + j * 16 + ccol;
#pragma unroll
            for (int r = 0; r < 8; ++r)
                C[rbase + (size_t)r * N_DIM + col] = acc[i][j][r];
        }
}

extern "C" void kernel_launch(void* const* d_in, const int* in_sizes, int n_in,
                              void* d_out, int out_size, void* d_ws, size_t ws_size,
                              hipStream_t stream) {
    const float* x = (const float*)d_in[0];      // [8,2048,4096] fp32
    const float* w = (const float*)d_in[1];      // [4096,4096]   fp32
    float* out = (float*)d_out;                  // [8,2048,4096] fp32

    char* ws = (char*)d_ws;
    float*     d_sum = (float*)ws;                                        // 4 B
    _Float16*  wq    = (_Float16*)(ws + 256);                             // 32 MB
    _Float16*  xh    = (_Float16*)(ws + 256 + (size_t)N_DIM * K_DIM * 2); // 128 MB

    hipMemsetAsync(d_sum, 0, sizeof(float), stream);

    bitnet_absum_kernel<<<2048, 256, 0, stream>>>(w, d_sum, N_DIM * K_DIM);

    {
        int n4 = (N_DIM * K_DIM) / 4;
        bitnet_quantw_kernel<<<(n4 + 255) / 256, 256, 0, stream>>>(w, d_sum, wq, n4);
    }
    {
        int n4 = (M_DIM * K_DIM) / 4;
        bitnet_cvtx_kernel<<<(n4 + 255) / 256, 256, 0, stream>>>(x, xh, n4);
    }

    dim3 grid(N_DIM / BN, M_DIM / BM);   // 32 x 128 blocks
    bitnet_wmma_gemm_kernel<<<grid, 128, 0, stream>>>(xh, wq, out);
}
